// Bidirectional_GRU_Decoder_19971597927081
// MI455X (gfx1250) — matrix operand order repaired and checked by reading, not compiled, not run on hardware
//
#include <hip/hip_runtime.h>

// ---------------------------------------------------------------------------
// Bidirectional 2-layer GRU decoder, persistent-RNN style for MI455X (gfx1250)
//   B=64, S=512, D=512, H=512.  4 GRU cells (0f,0b,1f,1b).
//   Matrix math: v_wmma_f32_16x16x32_bf16 (fp32 accumulate), weights LDS-resident.
// ---------------------------------------------------------------------------

#define Bsz 64
#define Ssz 512
#define Dsz 512
#define Hsz 512
#define NWG_PER_CELL 16          // each WG owns 32 hidden columns
#define NCOLS 32
#define NTILES 6                 // 3 gates x 2 subtiles of 16 cols
#define KBLK 16                  // K=512 / 32

typedef __attribute__((ext_vector_type(4)))  unsigned int v4u;
typedef __attribute__((ext_vector_type(8)))  float        v8f;
typedef __attribute__((ext_vector_type(16))) __bf16       v16bf;

union ABu { v4u q[2]; v16bf v; };

__device__ __forceinline__ unsigned short f2bf(float f) {
    unsigned int u = __float_as_uint(f);
    unsigned int r = u + 0x7fffu + ((u >> 16) & 1u);   // round-to-nearest-even
    return (unsigned short)(r >> 16);
}

// ---------------------------------------------------------------------------
// Prep kernel 1: zero sync counters
// ---------------------------------------------------------------------------
__global__ void reset_kernel(int* counters) {
    if (threadIdx.x < 4) counters[threadIdx.x] = 0;
}

// ---------------------------------------------------------------------------
// Prep kernel 2: x fp32 (B,S,D) -> bf16 (S,B,D)
// ---------------------------------------------------------------------------
__global__ void xconv_kernel(const float* __restrict__ x, unsigned short* __restrict__ xb) {
    int idx = blockIdx.x * 256 + threadIdx.x;            // 0 .. 16777215
    int b = idx >> 18;
    int rem = idx & 0x3FFFF;
    int t = rem >> 9;
    int d = rem & 511;
    xb[((size_t)t * Bsz + b) * Dsz + d] = f2bf(x[idx]);
}

// ---------------------------------------------------------------------------
// Prep kernel 3: pack weights into WMMA SRC1 (B-operand) lane order, bf16.
// Layout: [cell][wg][gemm][nt][kb][lane*16+i]  (gemm 0=Wih, 1=Whh)
// B, contiguous k order (same as A loads): lane<16 -> col N=lane, K = kb*32 + i ; lane>=16 -> K += 16.
// gru0_b consumes feature-reversed x -> reverse Wih0b columns here.
// ---------------------------------------------------------------------------
struct WP { const float* p[8]; };

__global__ void packw_kernel(WP wp, unsigned short* __restrict__ wpk) {
    int idx = blockIdx.x * 256 + threadIdx.x;            // 0 .. 6291455
    int r = idx;
    int cell = r / 1572864;  r %= 1572864;
    int wg   = r / 98304;    r %= 98304;
    int gemm = r / 49152;    r %= 49152;
    int nt   = r / 8192;     r %= 8192;
    int kb   = r / 512;      r %= 512;
    int lane = r >> 4;
    int i    = r & 15;
    int lc   = lane & 15;
    int hi   = lane >> 4;
    int gate = nt >> 1;
    int sub  = nt & 1;
    int nrow = gate * Hsz + wg * NCOLS + sub * 16 + lc;  // row of W (3H x K)
    int k    = kb * 32 + hi * 16 + i;
    if (cell == 1 && gemm == 0) k = 511 - k;             // feature flip for gru0_b
    const float* W = wp.p[cell * 2 + gemm];
    wpk[idx] = f2bf(W[nrow * 512 + k]);
}

// ---------------------------------------------------------------------------
// Prep kernel 4: seed bf16 hidden-state rings (slot 1 == "step -1") from enc_h
// ring layout: hring[(cell*2 + slot)*32768 + row*512 + col]
// ---------------------------------------------------------------------------
__global__ void inith_kernel(const float* __restrict__ enc_h, unsigned short* __restrict__ hring) {
    int idx = blockIdx.x * 256 + threadIdx.x;            // 0 .. 131071
    int cell = idx >> 15;
    int rem  = idx & 32767;
    int row  = rem >> 9;
    int col  = rem & 511;
    hring[(cell * 2 + 1) * 32768 + rem] = f2bf(enc_h[row * 1024 + (cell & 1) * 512 + col]);
}

// ---------------------------------------------------------------------------
// Persistent decoder kernel: 64 WGs = 4 cells x 16, 256 threads (8 wave32s)
// Wave mapping: g = gi/gh GEMM, mh = M half (2 tiles), ks = K half (8 kblocks)
// ---------------------------------------------------------------------------
struct DecParams {
    const unsigned short* xb;     // (S,B,D) bf16
    const unsigned short* wpk;    // packed weights bf16
    unsigned short*       hring;  // 4 cells x 2 slots x 64 x 512 bf16
    int*                  counters;
    const float*          bih[4];
    const float*          bhh[4];
    const float*          enc_h;
    float*                out;
};

__global__ __launch_bounds__(256, 1)
void decoder_kernel(DecParams P) {
    __shared__ v4u   s_w[2 * NTILES * KBLK * 64];        // 196608 B: Wih+Whh, resident
    __shared__ float s_acc[4 * 64 * 96];                 //  98304 B: (g,ks) partials
    __shared__ float s_bias[2 * 96];

    const int tid  = threadIdx.x;
    const int bid  = blockIdx.x;
    const int cell = bid >> 4;          // 0=0f 1=0b 2=1f 3=1b
    const int wg   = bid & 15;
    const int c0   = wg * NCOLS;
    const int isB  = cell & 1;
    const int wv   = tid >> 5;
    const int lane = tid & 31;
    const int g    = wv >> 2;           // 0: gi (Wih)   1: gh (Whh)
    const int mh   = (wv >> 1) & 1;     // M half: tiles {2mh, 2mh+1}
    const int ks   = wv & 1;            // K half: kblocks [8ks, 8ks+8)
    const int lc   = lane & 15;
    const int hi   = lane >> 4;

    // ---- preload BOTH packed weight GEMMs into LDS (contiguous slice) ----
    {
        const v4u* wsrc = (const v4u*)(P.wpk + ((size_t)(cell * 16 + wg)) * 2u * 49152u);
        for (int o = tid; o < 2 * NTILES * KBLK * 64; o += 256) s_w[o] = wsrc[o];
    }
    // biases -> LDS (constant across steps)
    if (tid < 192) {
        int gg = tid / 96, r96 = tid % 96;
        const float* bp = gg ? P.bhh[cell] : P.bih[cell];
        s_bias[tid] = bp[(r96 >> 5) * Hsz + c0 + (r96 & 31)];
    }
    // register-resident fp32 hidden state (8 elements / thread, fixed mapping)
    float hreg[8];
    for (int k = 0; k < 8; ++k) {
        int idx = tid + (k << 8);
        int row = idx >> 5, col = idx & 31;
        hreg[k] = P.enc_h[row * 1024 + isB * 512 + c0 + col];
    }
    __syncthreads();

    int* cnt = P.counters;
    const int NW = NWG_PER_CELL;

    for (int t = 0; t < Ssz; ++t) {
        // ---- cross-WG ordering (acquire) ----
        if (tid == 0) {
            int iA, iB, tgtA, tgtB;
            if      (cell == 0) { iA = 0; tgtA = NW * t;       iB = 2; tgtB = NW * (t - 1); }
            else if (cell == 1) { iA = 1; tgtA = NW * t;       iB = 3; tgtB = NW * (t - 1); }
            else if (cell == 2) { iA = 0; tgtA = NW * (t + 1); iB = 2; tgtB = NW * t; }
            else                { iA = 1; tgtA = NW * (t + 1); iB = 3; tgtB = NW * t; }
            if (tgtA > 0)
                while (__hip_atomic_load(&cnt[iA], __ATOMIC_ACQUIRE, __HIP_MEMORY_SCOPE_AGENT) < tgtA)
                    __builtin_amdgcn_s_sleep(1);
            if (tgtB > 0)
                while (__hip_atomic_load(&cnt[iB], __ATOMIC_ACQUIRE, __HIP_MEMORY_SCOPE_AGENT) < tgtB)
                    __builtin_amdgcn_s_sleep(1);
        }
        __syncthreads();

        // prefetch next timestep's input (layer-0 cells only)
        if (cell < 2 && t + 1 < Ssz) {
            __builtin_prefetch(P.xb + (size_t)(t + 1) * 32768 + tid * 64, 0, 1);
            __builtin_prefetch(P.xb + (size_t)(t + 1) * 32768 + 16384 + tid * 64, 0, 1);
        }

        // ---- A operand base (uniform select, no divergent control flow) ----
        const unsigned short* AbaseGi =
            (cell < 2) ? (P.xb + (size_t)t * 32768)
                       : (P.hring + ((cell - 2) * 2 + (t & 1)) * 32768);      // h0(t)
        const unsigned short* AbaseGh =
            P.hring + (cell * 2 + ((t + 1) & 1)) * 32768;                     // h(t-1)
        const unsigned short* Abase = g ? AbaseGh : AbaseGi;
        const unsigned short* arp0  = Abase + (size_t)((mh * 2 + 0) * 16 + lc) * 512;
        const unsigned short* arp1  = Abase + (size_t)((mh * 2 + 1) * 16 + lc) * 512;

        // ---- GEMM: 2 M-tiles x 6 N-tiles, K half = 8 kblocks, bf16 WMMA ----
        v8f acc0[NTILES] = {};
        v8f acc1[NTILES] = {};
        const int kb0 = ks * 8;
        for (int kk = 0; kk < 8; ++kk) {
            const int kb   = kb0 + kk;
            const int aoff = kb * 32 + hi * 16;    // A 16x32 lane layout
            ABu a0, a1;
            a0.q[0] = *(const v4u*)(arp0 + aoff);
            a0.q[1] = *(const v4u*)(arp0 + aoff + 8);
            a1.q[0] = *(const v4u*)(arp1 + aoff);
            a1.q[1] = *(const v4u*)(arp1 + aoff + 8);
            ABu b[NTILES];
            const int bbase = g * (NTILES * KBLK * 64) + kb * 64 + lane * 2;
            #pragma unroll
            for (int nt = 0; nt < NTILES; ++nt) {
                b[nt].q[0] = s_w[bbase + nt * (KBLK * 64)];
                b[nt].q[1] = s_w[bbase + nt * (KBLK * 64) + 1];
            }
            #pragma unroll
            for (int nt = 0; nt < NTILES; ++nt) {
                acc0[nt] = __builtin_amdgcn_wmma_f32_16x16x32_bf16(
                    false, a0.v, false, b[nt].v, (short)0, acc0[nt], false, false);
                acc1[nt] = __builtin_amdgcn_wmma_f32_16x16x32_bf16(
                    false, a1.v, false, b[nt].v, (short)0, acc1[nt], false, false);
            }
        }

        // ---- spill K-partial accumulators to LDS: bank (g*2+ks) ----
        {
            const int gk = g * 2 + ks;
            #pragma unroll
            for (int nt = 0; nt < NTILES; ++nt)
                #pragma unroll
                for (int r = 0; r < 8; ++r) {
                    int row0 = (mh * 2 + 0) * 16 + hi * 8 + r;
                    int row1 = (mh * 2 + 1) * 16 + hi * 8 + r;
                    s_acc[(gk * 64 + row0) * 96 + nt * 16 + lc] = acc0[nt][r];
                    s_acc[(gk * 64 + row1) * 96 + nt * 16 + lc] = acc1[nt][r];
                }
        }
        __syncthreads();

        // ---- fused GRU gates: 8 (row,col) elements / thread ----
        for (int k = 0; k < 8; ++k) {
            int idx = tid + (k << 8);
            int row = idx >> 5, col = idx & 31;
            const float* A0 = &s_acc[(0 * 64 + row) * 96];   // gi, K half 0
            const float* A1 = &s_acc[(1 * 64 + row) * 96];   // gi, K half 1
            const float* A2 = &s_acc[(2 * 64 + row) * 96];   // gh, K half 0
            const float* A3 = &s_acc[(3 * 64 + row) * 96];   // gh, K half 1
            float gi_r = A0[col]      + A1[col]      + s_bias[col];
            float gi_z = A0[32 + col] + A1[32 + col] + s_bias[32 + col];
            float gi_n = A0[64 + col] + A1[64 + col] + s_bias[64 + col];
            float gh_r = A2[col]      + A3[col]      + s_bias[96 + col];
            float gh_z = A2[32 + col] + A3[32 + col] + s_bias[96 + 32 + col];
            float gh_n = A2[64 + col] + A3[64 + col] + s_bias[96 + 64 + col];
            float rr = 1.f / (1.f + __expf(-(gi_r + gh_r)));
            float zz = 1.f / (1.f + __expf(-(gi_z + gh_z)));
            float nn = tanhf(gi_n + rr * gh_n);
            float h  = (1.f - zz) * nn + zz * hreg[k];
            hreg[k] = h;
            P.hring[(cell * 2 + (t & 1)) * 32768 + row * 512 + c0 + col] = f2bf(h);
            if (cell >= 2) {
                size_t o = ((size_t)row * Ssz + t) * 1024 + (size_t)isB * 512 + c0 + col;
                P.out[o] = h;
                if (t == Ssz - 1)
                    P.out[(size_t)Bsz * Ssz * 1024 + row * 1024 + isB * 512 + c0 + col] = h;
            }
        }
        __threadfence();
        __syncthreads();
        if (tid == 0)
            __hip_atomic_fetch_add(&cnt[cell], 1, __ATOMIC_RELEASE, __HIP_MEMORY_SCOPE_AGENT);
    }
}

// ---------------------------------------------------------------------------
// Host launch
// ---------------------------------------------------------------------------
extern "C" void kernel_launch(void* const* d_in, const int* in_sizes, int n_in,
                              void* d_out, int out_size, void* d_ws, size_t ws_size,
                              hipStream_t stream) {
    (void)in_sizes; (void)n_in; (void)out_size; (void)ws_size;

    const float* input = (const float*)d_in[0];
    const float* enc_h = (const float*)d_in[2];

    char* ws = (char*)d_ws;
    int*            counters = (int*)ws;
    unsigned short* xb    = (unsigned short*)(ws + 256);
    unsigned short* wpk   = (unsigned short*)(ws + 256 + 33554432);
    unsigned short* hring = (unsigned short*)(ws + 256 + 33554432 + 12582912);

    reset_kernel<<<1, 64, 0, stream>>>(counters);
    xconv_kernel<<<65536, 256, 0, stream>>>(input, xb);

    WP wp;
    for (int c = 0; c < 4; ++c) {
        wp.p[c * 2 + 0] = (const float*)d_in[3 + c * 4 + 0];   // Wih
        wp.p[c * 2 + 1] = (const float*)d_in[3 + c * 4 + 1];   // Whh
    }
    packw_kernel<<<24576, 256, 0, stream>>>(wp, wpk);
    inith_kernel<<<512, 256, 0, stream>>>(enc_h, hring);

    DecParams P;
    P.xb = xb; P.wpk = wpk; P.hring = hring; P.counters = counters;
    for (int c = 0; c < 4; ++c) {
        P.bih[c] = (const float*)d_in[3 + c * 4 + 2];
        P.bhh[c] = (const float*)d_in[3 + c * 4 + 3];
    }
    P.enc_h = enc_h;
    P.out   = (float*)d_out;

    decoder_kernel<<<64, 256, 0, stream>>>(P);
}
